// VectorPrefixEncoder_85959475462367
// MI455X (gfx1250) — matrix operand order repaired and checked by reading, not compiled, not run on hardware
//
#include <hip/hip_runtime.h>
#include <hip/hip_bf16.h>
#include <stdint.h>

// ---------------------------------------------------------------------------
// VectorPrefixEncoder forward on gfx1250 (MI455X).
// All GEMM-shaped math on v_wmma_f32_16x16x32_bf16. Activations/weights are
// converted to bf16 once and streamed; GEMM inner loop = pure b128 loads+WMMA.
// ---------------------------------------------------------------------------

#define BB    4
#define LL    2048
#define DDIM  256
#define HH    8
#define FFD   1024
#define NLAYER 3
#define BL    (BB*LL)            // 8192 tokens
#define BLD   (BL*DDIM)          // 2,097,152

typedef __attribute__((ext_vector_type(16))) __bf16 v16bf;
typedef __attribute__((ext_vector_type(8)))  __bf16 v8bf;
typedef __attribute__((ext_vector_type(8)))  float  v8f;

static __device__ __forceinline__ __bf16 f2bf(float f) {
    // round-to-nearest-even f32 -> bf16 (codegen-safe, epilogue-only)
    unsigned u = __builtin_bit_cast(unsigned, f);
    u += 0x7FFFu + ((u >> 16) & 1u);
    unsigned short s = (unsigned short)(u >> 16);
    return __builtin_bit_cast(__bf16, s);
}

static __device__ __forceinline__ v16bf cat8(v8bf a, v8bf b) {
    return __builtin_shufflevector(a, b, 0,1,2,3,4,5,6,7,8,9,10,11,12,13,14,15);
}

static __device__ __forceinline__ float gelu_exact(float x) {
    return 0.5f * x * (1.0f + erff(x * 0.7071067811865476f));
}

// ---------------------------------------------------------------------------
// Embedding front-end + input LayerNorm. One block (256 thr) per token.
// ---------------------------------------------------------------------------
__global__ void __launch_bounds__(256)
k_embed(const float* __restrict__ seq_arg, const int* __restrict__ seq_cmd,
        const uint8_t* __restrict__ seq_mask, const int* __restrict__ contour_ids,
        const int* __restrict__ seq_topo,
        const float* __restrict__ emb_cmd, const float* __restrict__ w_cmd, const float* __restrict__ b_cmd,
        const float* __restrict__ w_arg1, const float* __restrict__ b_arg1,
        const float* __restrict__ w_arg2, const float* __restrict__ b_arg2,
        const float* __restrict__ w_argp, const float* __restrict__ b_argp,
        const float* __restrict__ w_pabs, const float* __restrict__ b_pabs,
        const float* __restrict__ w_prel, const float* __restrict__ b_prel,
        const float* __restrict__ emb_cid, const float* __restrict__ w_cid, const float* __restrict__ b_cid,
        const float* __restrict__ w_topo, const float* __restrict__ b_topo,
        const float* __restrict__ g_in, const float* __restrict__ bln_in,
        float* __restrict__ xout)
{
    __shared__ float s_ecmd[128], s_a1[128], s_a2[128], s_pe[64], s_ecid[64], s_topo[3], s_red[256];
    const int token = blockIdx.x;
    const int l = token % LL;
    const int t = threadIdx.x;

    const int cmd = seq_cmd[token];
    const uint8_t mk = seq_mask[token];
    const int cid = max(contour_ids[token], 0);
    const int tp  = seq_topo[token];
    const bool ml = mk && (cmd == 1 || cmd == 2);
    const bool mq = mk && (cmd == 3);
    const float g01 = (ml || mq) ? 1.f : 0.f;
    const float g23 = mq ? 1.f : 0.f;
    const float a0 = seq_arg[token*4+0]*g01, a1v = seq_arg[token*4+1]*g01;
    const float a2v = seq_arg[token*4+2]*g23, a3v = seq_arg[token*4+3]*g23;

    if (t < 128) {
        s_ecmd[t] = emb_cmd[cmd*128 + t];
        float z = a0*w_arg1[t] + a1v*w_arg1[128+t] + a2v*w_arg1[256+t] + a3v*w_arg1[384+t] + b_arg1[t];
        s_a1[t] = gelu_exact(z);
    }
    if (t < 64) s_ecid[t] = emb_cid[cid*64 + t];
    if (t < 32) {
        const float div = __expf(-(float)(2*t) * 0.29710775393471563f); // ln(1e4)/31
        const float ang = (float)l * div;
        s_pe[2*t]   = sinf(ang);
        s_pe[2*t+1] = cosf(ang);
    }
    if (t < 3) s_topo[t] = ((tp >> t) & 1) ? 1.f : 0.f;
    __syncthreads();

    if (t < 128) {
        float z = b_arg2[t];
        #pragma unroll 4
        for (int k = 0; k < 128; ++k) z += s_a1[k] * w_arg2[k*128 + t];
        s_a2[t] = z;
    }
    __syncthreads();

    float acc = b_cmd[t] + b_argp[t] + b_pabs[t] + b_prel[t] + b_cid[t] + b_topo[t];
    #pragma unroll 4
    for (int k = 0; k < 128; ++k)
        acc += s_ecmd[k]*w_cmd[k*DDIM + t] + s_a2[k]*w_argp[k*DDIM + t];
    #pragma unroll 4
    for (int k = 0; k < 64; ++k)
        acc += s_pe[k]*(w_pabs[k*DDIM + t] + w_prel[k*DDIM + t]) + s_ecid[k]*w_cid[k*DDIM + t];
    for (int k = 0; k < 3; ++k) acc += s_topo[k]*w_topo[k*DDIM + t];

    s_red[t] = acc; __syncthreads();
    for (int s = 128; s > 0; s >>= 1) { if (t < s) s_red[t] += s_red[t+s]; __syncthreads(); }
    const float mean = s_red[0] * (1.f/DDIM);
    __syncthreads();
    const float dm = acc - mean;
    s_red[t] = dm*dm; __syncthreads();
    for (int s = 128; s > 0; s >>= 1) { if (t < s) s_red[t] += s_red[t+s]; __syncthreads(); }
    const float var = s_red[0] * (1.f/DDIM);
    xout[(size_t)token*DDIM + t] = dm * rsqrtf(var + 1e-5f) * g_in[t] + bln_in[t];
}

// ---------------------------------------------------------------------------
// LayerNorm: f32 in, bf16 out (feeds WMMA A operand). One block per token.
// ---------------------------------------------------------------------------
__global__ void __launch_bounds__(256)
k_layernorm(const float* __restrict__ x, const float* __restrict__ g,
            const float* __restrict__ bt, __bf16* __restrict__ out)
{
    __shared__ float sr[256];
    const int tok = blockIdx.x;
    const int t = threadIdx.x;
    const float v = x[(size_t)tok*DDIM + t];
    sr[t] = v; __syncthreads();
    for (int s = 128; s > 0; s >>= 1) { if (t < s) sr[t] += sr[t+s]; __syncthreads(); }
    const float mean = sr[0] * (1.f/DDIM);
    __syncthreads();
    const float d = v - mean;
    sr[t] = d*d; __syncthreads();
    for (int s = 128; s > 0; s >>= 1) { if (t < s) sr[t] += sr[t+s]; __syncthreads(); }
    const float var = sr[0] * (1.f/DDIM);
    out[(size_t)tok*DDIM + t] = f2bf(d * rsqrtf(var + 1e-5f) * g[t] + bt[t]);
}

// ---------------------------------------------------------------------------
// Weight repack: f32 [K,N] -> bf16 in B-fragment order.
// grid = (K/32, N/16), block = 32. Tile (kt,ct): lane holds 16 contiguous bf16
// = pairs (W[kb,c], W[kb+1,c]) with kb = kt*32 + 16*(lane/16) + 2v, c = ct*16+lane%16.
// ---------------------------------------------------------------------------
__global__ void __launch_bounds__(32)
k_repack(const float* __restrict__ W, __bf16* __restrict__ Wp, int N)
{
    const int lane = threadIdx.x;
    const int hi = lane >> 4, lo = lane & 15;
    const int k0 = blockIdx.x * 32;
    const int c  = blockIdx.y * 16 + lo;
    __bf16* out = Wp + (((size_t)blockIdx.x * gridDim.y + blockIdx.y) * 32 + lane) * 16;
    #pragma unroll
    for (int v = 0; v < 8; ++v) {
        const int kb = k0 + 16*hi + 2*v;
        out[2*v]   = f2bf(W[(size_t)kb*N + c]);
        out[2*v+1] = f2bf(W[(size_t)(kb+1)*N + c]);
    }
}

// ---------------------------------------------------------------------------
// WMMA GEMM: C[M,N] = act(A[M,K]bf16 @ Wp + bias) (+Res f32).
// grid = (N/32, M/64), block = 128. Wave computes 16x32 (A frag reused twice).
// Inner loop: 2x b128 A loads + 4x b128 B loads + 2 WMMA. Zero VALU.
// ---------------------------------------------------------------------------
__global__ void __launch_bounds__(128)
k_gemm(const __bf16* __restrict__ A, const __bf16* __restrict__ Wp,
       const float* __restrict__ bias, void* __restrict__ Cout,
       const float* __restrict__ Res, int M, int N, int K,
       int do_gelu, int out_bf16)
{
    const int lane = threadIdx.x & 31;
    const int wave = threadIdx.x >> 5;
    const int hi = lane >> 4, lo = lane & 15;
    const int rowBase = blockIdx.y*64 + wave*16;
    const int ct0 = blockIdx.x * 2;            // two 16-col tiles
    const int ntile = N >> 4;

    v8f acc0 = {}, acc1 = {};
    for (int k0 = 0; k0 < K; k0 += 32) {
        const __bf16* ar = A + (size_t)(rowBase+lo)*K + k0 + 16*hi;
        const v16bf a = cat8(*(const v8bf*)ar, *(const v8bf*)(ar + 8));

        const size_t tbase = ((size_t)(k0 >> 5) * ntile + ct0) * 32;
        const __bf16* w0 = Wp + (tbase + lane) * 16;
        const __bf16* w1 = Wp + (tbase + 32 + lane) * 16;
        const v16bf b0 = cat8(*(const v8bf*)w0, *(const v8bf*)(w0 + 8));
        const v16bf b1 = cat8(*(const v8bf*)w1, *(const v8bf*)(w1 + 8));

        acc0 = __builtin_amdgcn_wmma_f32_16x16x32_bf16(false, a, false, b0, (short)0, acc0, false, false);
        acc1 = __builtin_amdgcn_wmma_f32_16x16x32_bf16(false, a, false, b1, (short)0, acc1, false, false);
    }

    #pragma unroll
    for (int g = 0; g < 2; ++g) {
        const v8f* accp = g ? &acc1 : &acc0;
        const int c = (ct0 + g)*16 + lo;
        #pragma unroll
        for (int v = 0; v < 8; ++v) {
            const int r = rowBase + v + 8*hi;
            float val = (*accp)[v] + bias[c];
            if (do_gelu) val = gelu_exact(val);
            if (out_bf16) {
                ((__bf16*)Cout)[(size_t)r*N + c] = f2bf(val);
            } else {
                ((float*)Cout)[(size_t)r*N + c] = val + Res[(size_t)r*N + c];
            }
        }
    }
}

// ---------------------------------------------------------------------------
// V transpose: qkv v-slice [token, d] -> vT[b, h, d%32, l] (keys contiguous).
// ---------------------------------------------------------------------------
__global__ void __launch_bounds__(256)
k_vtrans(const __bf16* __restrict__ qkvb, __bf16* __restrict__ vT)
{
    const int t = blockIdx.x;
    const int d = threadIdx.x;
    const int b = t / LL, l = t % LL;
    const __bf16 val = qkvb[(size_t)t*768 + 512 + d];
    const int h = d >> 5, dd = d & 31;
    vT[((size_t)(b*HH + h)*32 + dd)*LL + l] = val;
}

// ---------------------------------------------------------------------------
// Flash attention: one wave per 16-query tile per (b,h).
// grid = (L/64, H, B), block = 128. All fragments are contiguous b128 loads.
// ---------------------------------------------------------------------------
__global__ void __launch_bounds__(128)
k_attn(const __bf16* __restrict__ qkv, const __bf16* __restrict__ vT,
       const uint8_t* __restrict__ mask, __bf16* __restrict__ y)
{
    __shared__ __bf16 sPb[4][16*32];
    const int lane = threadIdx.x & 31;
    const int wave = threadIdx.x >> 5;
    const int hi = lane >> 4, lo = lane & 15;
    const int qbase = blockIdx.x*64 + wave*16;
    const int h = blockIdx.y;
    const int b = blockIdx.z;
    const float scale = 0.17677669529663687f;   // 1/sqrt(32)
    const __bf16* base = qkv + (size_t)b*LL*768;
    const uint8_t* mb = mask + (size_t)b*LL;

    // Q fragment (A matrix 16x32 over dh)
    const __bf16* qr = base + (size_t)(qbase+lo)*768 + h*32 + 16*hi;
    const v16bf aq = cat8(*(const v8bf*)qr, *(const v8bf*)(qr + 8));

    float mrow[8], lrow[8];
    v8f o0 = {}, o1 = {};
    #pragma unroll
    for (int v = 0; v < 8; ++v) { mrow[v] = -1e30f; lrow[v] = 0.f; }

    for (int kt = 0; kt < LL; kt += 32) {
        v8f s0, s1;
        #pragma unroll
        for (int half = 0; half < 2; ++half) {
            const int jb = kt + 16*half;
            // K fragment (B matrix 32x16): 16 contiguous bf16 per lane
            const __bf16* kr = base + (size_t)(jb+lo)*768 + 256 + h*32 + 16*hi;
            const v16bf bk = cat8(*(const v8bf*)kr, *(const v8bf*)(kr + 8));
            v8f z = {};
            v8f s = __builtin_amdgcn_wmma_f32_16x16x32_bf16(false, aq, false, bk,
                                                            (short)0, z, false, false);
            const float madd = mb[jb+lo] ? 0.f : -1e30f;   // key-padding mask
            #pragma unroll
            for (int v = 0; v < 8; ++v) s[v] = s[v]*scale + madd;
            if (half == 0) s0 = s; else s1 = s;
        }

        // streaming-softmax update (row = v + 8*hi, spread across 16 lanes)
        float p0[8], p1[8];
        #pragma unroll
        for (int v = 0; v < 8; ++v) {
            float mx = fmaxf(s0[v], s1[v]);
            #pragma unroll
            for (int off = 1; off < 16; off <<= 1) mx = fmaxf(mx, __shfl_xor(mx, off, 16));
            const float mnew = fmaxf(mrow[v], mx);
            const float alpha = __expf(mrow[v] - mnew);
            p0[v] = __expf(s0[v] - mnew);
            p1[v] = __expf(s1[v] - mnew);
            float rs = p0[v] + p1[v];
            #pragma unroll
            for (int off = 1; off < 16; off <<= 1) rs += __shfl_xor(rs, off, 16);
            lrow[v] = lrow[v]*alpha + rs;
            mrow[v] = mnew;
            o0[v] *= alpha;
            o1[v] *= alpha;
        }

        // re-layout P (C layout) -> A fragment via bf16 LDS
        __syncthreads();
        #pragma unroll
        for (int v = 0; v < 8; ++v) {
            const int r = v + 8*hi;
            sPb[wave][r*32 + lo]      = f2bf(p0[v]);
            sPb[wave][r*32 + 16 + lo] = f2bf(p1[v]);
        }
        __syncthreads();
        const __bf16* pr = &sPb[wave][lo*32 + 16*hi];
        const v16bf ap = cat8(*(const v8bf*)pr, *(const v8bf*)(pr + 8));

        // P(16x32) @ V(32x32): V fragments contiguous in vT
        #pragma unroll
        for (int nt = 0; nt < 2; ++nt) {
            const __bf16* vr = vT + ((size_t)(b*HH + h)*32 + nt*16 + lo)*LL + kt + 16*hi;
            const v16bf bv = cat8(*(const v8bf*)vr, *(const v8bf*)(vr + 8));
            if (nt == 0)
                o0 = __builtin_amdgcn_wmma_f32_16x16x32_bf16(false, ap, false, bv, (short)0, o0, false, false);
            else
                o1 = __builtin_amdgcn_wmma_f32_16x16x32_bf16(false, ap, false, bv, (short)0, o1, false, false);
        }
    }

    #pragma unroll
    for (int v = 0; v < 8; ++v) {
        const int r = qbase + v + 8*hi;
        const float inv = 1.f / lrow[v];
        __bf16* yp = y + (size_t)(b*LL + r)*DDIM + h*32;
        yp[lo]      = f2bf(o0[v]*inv);
        yp[16 + lo] = f2bf(o1[v]*inv);
    }
}

// ---------------------------------------------------------------------------
// Finalize: copy x -> out, write kpm as float (1.0 = padding).
// ---------------------------------------------------------------------------
__global__ void __launch_bounds__(256)
k_finalize(const float* __restrict__ x, const uint8_t* __restrict__ mask,
           float* __restrict__ out)
{
    const int i = blockIdx.x*256 + threadIdx.x;
    out[i] = x[i];
    if (i < BB*LL) out[BLD + BB*DDIM + i] = mask[i] ? 0.f : 1.f;
}

// ---------------------------------------------------------------------------
// Masked mean pooling -> prefix_repr [B, D].
// ---------------------------------------------------------------------------
__global__ void __launch_bounds__(256)
k_prefix(const float* __restrict__ x, const uint8_t* __restrict__ mask,
         float* __restrict__ out)
{
    const int b = blockIdx.x, d = threadIdx.x;
    float s = 0.f, cnt = 0.f;
    for (int l = 0; l < LL; ++l) {
        const float m = mask[b*LL + l] ? 1.f : 0.f;
        s += x[((size_t)b*LL + l)*DDIM + d] * m;
        cnt += m;
    }
    out[BLD + b*DDIM + d] = s / fmaxf(cnt, 1.f);
}

// ---------------------------------------------------------------------------
// Launcher. d_in assumes dict-insertion-order flattening of setup_inputs().
// ---------------------------------------------------------------------------
extern "C" void kernel_launch(void* const* d_in, const int* in_sizes, int n_in,
                              void* d_out, int out_size, void* d_ws, size_t ws_size,
                              hipStream_t stream)
{
    (void)in_sizes; (void)n_in; (void)out_size; (void)ws_size;

    const float* seq_arg  = (const float*)d_in[0];
    const float* emb_cmd  = (const float*)d_in[1];
    const float* w_cmd    = (const float*)d_in[2];
    const float* b_cmd    = (const float*)d_in[3];
    const float* w_arg1   = (const float*)d_in[4];
    const float* b_arg1   = (const float*)d_in[5];
    const float* w_arg2   = (const float*)d_in[6];
    const float* b_arg2   = (const float*)d_in[7];
    const float* w_argp   = (const float*)d_in[8];
    const float* b_argp   = (const float*)d_in[9];
    const float* w_pabs   = (const float*)d_in[10];
    const float* b_pabs   = (const float*)d_in[11];
    const float* w_prel   = (const float*)d_in[12];
    const float* b_prel   = (const float*)d_in[13];
    const float* emb_cid  = (const float*)d_in[14];
    const float* w_cid    = (const float*)d_in[15];
    const float* b_cid    = (const float*)d_in[16];
    const float* w_topo   = (const float*)d_in[17];
    const float* b_topo   = (const float*)d_in[18];
    const float* ln_in_g  = (const float*)d_in[19];
    const float* ln_in_b  = (const float*)d_in[20];
    const float* ln1_g    = (const float*)d_in[21];
    const float* ln1_b    = (const float*)d_in[22];
    const float* w_qkv    = (const float*)d_in[23];
    const float* b_qkv    = (const float*)d_in[24];
    const float* w_o      = (const float*)d_in[25];
    const float* b_o      = (const float*)d_in[26];
    const float* ln2_g    = (const float*)d_in[27];
    const float* ln2_b    = (const float*)d_in[28];
    const float* w_f1     = (const float*)d_in[29];
    const float* b_f1     = (const float*)d_in[30];
    const float* w_f2     = (const float*)d_in[31];
    const float* b_f2     = (const float*)d_in[32];
    const int*     seq_cmd     = (const int*)d_in[33];
    const uint8_t* seq_mask    = (const uint8_t*)d_in[34];
    const int*     contour_ids = (const int*)d_in[35];
    const int*     seq_topo    = (const int*)d_in[36];

    float*  xbuf = (float*)d_ws;                       // [BL, D] f32
    __bf16* hbuf = (__bf16*)(xbuf + (size_t)BLD);      // [BL, D] bf16
    __bf16* qkvb = hbuf + (size_t)BLD;                 // [BL, 3D] bf16
    __bf16* vTb  = qkvb + (size_t)BL*768;              // [B,H,32,L] bf16
    __bf16* ybuf = vTb  + (size_t)BLD;                 // [BL, D] bf16
    __bf16* ubuf = ybuf + (size_t)BLD;                 // [BL, FF] bf16
    __bf16* wpk  = ubuf + (size_t)BL*FFD;              // repacked weights
    const size_t WL = 786432;                          // per-layer repacked elems
    const size_t OFF_QKV = 0, OFF_O = 196608, OFF_F1 = 262144, OFF_F2 = 524288;

    // repack all weights to WMMA B-fragment order (bf16)
    for (int i = 0; i < NLAYER; ++i) {
        __bf16* wl = wpk + (size_t)i*WL;
        k_repack<<<dim3(DDIM/32, 768/16), 32, 0, stream>>>(w_qkv + (size_t)i*DDIM*768, wl + OFF_QKV, 768);
        k_repack<<<dim3(DDIM/32, DDIM/16), 32, 0, stream>>>(w_o   + (size_t)i*DDIM*DDIM, wl + OFF_O, DDIM);
        k_repack<<<dim3(DDIM/32, FFD/16),  32, 0, stream>>>(w_f1  + (size_t)i*DDIM*FFD, wl + OFF_F1, FFD);
        k_repack<<<dim3(FFD/32,  DDIM/16), 32, 0, stream>>>(w_f2  + (size_t)i*FFD*DDIM, wl + OFF_F2, DDIM);
    }

    k_embed<<<BL, 256, 0, stream>>>(seq_arg, seq_cmd, seq_mask, contour_ids, seq_topo,
                                    emb_cmd, w_cmd, b_cmd, w_arg1, b_arg1, w_arg2, b_arg2,
                                    w_argp, b_argp, w_pabs, b_pabs, w_prel, b_prel,
                                    emb_cid, w_cid, b_cid, w_topo, b_topo,
                                    ln_in_g, ln_in_b, xbuf);

    for (int i = 0; i < NLAYER; ++i) {
        __bf16* wl = wpk + (size_t)i*WL;
        k_layernorm<<<BL, 256, 0, stream>>>(xbuf, ln1_g + i*DDIM, ln1_b + i*DDIM, hbuf);
        {   dim3 g(768/32, BL/64);
            k_gemm<<<g, 128, 0, stream>>>(hbuf, wl + OFF_QKV, b_qkv + i*768,
                                          qkvb, nullptr, BL, 768, DDIM, 0, 1); }
        k_vtrans<<<BL, 256, 0, stream>>>(qkvb, vTb);
        {   dim3 g(LL/64, HH, BB);
            k_attn<<<g, 128, 0, stream>>>(qkvb, vTb, seq_mask, ybuf); }
        {   dim3 g(DDIM/32, BL/64);
            k_gemm<<<g, 128, 0, stream>>>(ybuf, wl + OFF_O, b_o + i*DDIM,
                                          xbuf, xbuf, BL, DDIM, DDIM, 0, 0); }
        k_layernorm<<<BL, 256, 0, stream>>>(xbuf, ln2_g + i*DDIM, ln2_b + i*DDIM, hbuf);
        {   dim3 g(FFD/32, BL/64);
            k_gemm<<<g, 128, 0, stream>>>(hbuf, wl + OFF_F1, b_f1 + i*FFD,
                                          ubuf, nullptr, BL, FFD, DDIM, 1, 1); }
        {   dim3 g(DDIM/32, BL/64);
            k_gemm<<<g, 128, 0, stream>>>(ubuf, wl + OFF_F2, b_f2 + i*DDIM,
                                          xbuf, xbuf, BL, DDIM, FFD, 0, 0); }
    }

    k_finalize<<<BLD/256, 256, 0, stream>>>(xbuf, seq_mask, (float*)d_out);
    k_prefix<<<BB, 256, 0, stream>>>(xbuf, seq_mask, (float*)d_out);
}
